// MetaPathTeacher_60284160966707
// MI455X (gfx1250) — matrix operand order repaired and checked by reading, not compiled, not run on hardware
//
#include <hip/hip_runtime.h>
#include <hip/hip_bf16.h>
#include <math.h>

#define N_NODES 20000
#define IN_DIM  512
#define D_HID   256
#define HEADS   8
#define DH      32
#define P_TYPES 4
#define E_EDGES 320000
#define CLASSES 16
#define NEG_SLOPE 0.2f

typedef __attribute__((ext_vector_type(16))) __bf16 v16bf;
typedef __attribute__((ext_vector_type(8)))  __bf16 v8bf;
typedef __attribute__((ext_vector_type(8)))  float  v8f;

// Native fire-and-forget f32 VMEM atomic (global_atomic_add_f32).
__device__ __forceinline__ void atomic_add_f32(float* p, float v) {
    unsafeAtomicAdd(p, v);
}

// ---------------- utility kernels ----------------

__global__ void k_zero_f32(float* __restrict__ p, int n) {
    int i = blockIdx.x * 256 + threadIdx.x;
    if (i < n) p[i] = 0.f;
}

__global__ void k_f32_to_bf16(const float* __restrict__ s, __bf16* __restrict__ d, int n) {
    int i = blockIdx.x * 256 + threadIdx.x;
    if (i < n) d[i] = (__bf16)s[i];
}

// W [K,Nc] f32 row-major  ->  Wt [Nc,K] bf16 row-major (B^T for WMMA B-operand)
__global__ void k_transpose_to_bf16(const float* __restrict__ W, __bf16* __restrict__ Wt,
                                    int K, int Nc) {
    int i = blockIdx.x * 256 + threadIdx.x;
    if (i < K * Nc) {
        int k = i / Nc, c = i % Nc;
        Wt[(size_t)c * K + k] = (__bf16)W[i];
    }
}

// ---------------- shared WMMA panel loop ----------------
// One wave computes a 16 x (16*NT) tile of A[M,K] * Bt[Nc,K]^T.
// A layout per ISA: half-wave K-base +0/+8; elems 0-7 -> K=kb..kb+7, elems 8-15 -> K=kb+16..kb+23
// B layout (matches A): half-wave K-base +0/+8; elems 0-7 -> K=kb..kb+7, elems 8-15 -> K=kb+16..kb+23
template<int NT>
__device__ __forceinline__ void wmma_panel(const __bf16* __restrict__ arow,
                                           const __bf16* __restrict__ brow,
                                           int K, int half, v8f acc[NT]) {
    for (int k0 = 0; k0 < K; k0 += 32) {
        const int ka = k0 + half * 8;
        const int kb = k0 + half * 8;
        v8bf alo = *(const v8bf*)(arow + ka);
        v8bf ahi = *(const v8bf*)(arow + ka + 16);
        v16bf a;
#pragma unroll
        for (int i = 0; i < 8; i++) { a[i] = alo[i]; a[i + 8] = ahi[i]; }
#pragma unroll
        for (int t = 0; t < NT; t++) {
            const __bf16* bp = brow + (size_t)t * 16 * K;
            v8bf blo = *(const v8bf*)(bp + kb);
            v8bf bhi = *(const v8bf*)(bp + kb + 16);
            v16bf b;
#pragma unroll
            for (int i = 0; i < 8; i++) { b[i] = blo[i]; b[i + 8] = bhi[i]; }
            acc[t] = __builtin_amdgcn_wmma_f32_16x16x32_bf16(false, a, false, b,
                                                             (short)0, acc[t], false, false);
        }
    }
}

// ---------------- WMMA GEMM: C[M,Nc] = A @ Bt^T + bias ----------------
template<int NT>
__global__ __launch_bounds__(32)
void k_wmma_gemm_bias(const __bf16* __restrict__ A, const __bf16* __restrict__ Bt,
                      const float* __restrict__ bias, float* __restrict__ C,
                      int K, int Nc) {
    const int m0   = blockIdx.x << 4;
    const int n0   = blockIdx.y * (16 * NT);
    const int lane = threadIdx.x;
    const int half = lane >> 4;
    const int l15  = lane & 15;
    const __bf16* arow = A  + (size_t)(m0 + l15) * K;
    const __bf16* brow = Bt + (size_t)(n0 + l15) * K;
    v8f acc[NT];
#pragma unroll
    for (int t = 0; t < NT; t++) acc[t] = (v8f){};
    wmma_panel<NT>(arow, brow, K, half, acc);
#pragma unroll
    for (int t = 0; t < NT; t++) {
        const int col = n0 + t * 16 + l15;
        const float bv = bias[col];
        float* crow = C + (size_t)(m0 + half * 8) * Nc + col;
#pragma unroll
        for (int v = 0; v < 8; v++) crow[(size_t)v * Nc] = acc[t][v] + bv;
    }
}

// ---------------- per-head attention coefficients ----------------
__global__ void k_head_scores(const float* __restrict__ z, const float* __restrict__ att_s,
                              const float* __restrict__ att_d,
                              float* __restrict__ a_src, float* __restrict__ a_dst) {
    int idx = blockIdx.x * 256 + threadIdx.x;
    if (idx >= N_NODES * HEADS) return;
    int node = idx >> 3, h = idx & 7;
    const float* zp = z + (size_t)node * D_HID + h * DH;
    const float* ws = att_s + h * DH;
    const float* wd = att_d + h * DH;
    float ss = 0.f, sd = 0.f;
#pragma unroll
    for (int d = 0; d < DH; d++) { float zv = zp[d]; ss += zv * ws[d]; sd += zv * wd[d]; }
    a_src[idx] = ss;
    a_dst[idx] = sd;
}

// ---------------- edge pass 1: leaky_relu + exp + denom scatter ----------------
__global__ void k_edge_softmax_denom(const int* __restrict__ src, const int* __restrict__ dst,
                                     const float* __restrict__ a_src, const float* __restrict__ a_dst,
                                     float* __restrict__ exbuf, float* __restrict__ denom) {
    int i = blockIdx.x * 256 + threadIdx.x;
    if (i >= E_EDGES) return;
    int s = src[i], d = dst[i];
    const float* as = a_src + (size_t)s * HEADS;
    const float* ad = a_dst + (size_t)d * HEADS;
    float* ex = exbuf + (size_t)i * HEADS;
    float* dn = denom + (size_t)d * HEADS;
#pragma unroll
    for (int h = 0; h < HEADS; h++) {
        float e = as[h] + ad[h];
        e = (e > 0.f) ? e : NEG_SLOPE * e;
        float x = __expf(e);
        ex[h] = x;
        atomic_add_f32(dn + h, x);
    }
}

// ---------------- edge pass 2: alpha-weighted z scatter (one wave per edge) ----------------
__global__ void k_edge_scatter(const int* __restrict__ src, const int* __restrict__ dst,
                               const float* __restrict__ z, const float* __restrict__ exbuf,
                               const float* __restrict__ denom, float* __restrict__ acc) {
    int gid  = blockIdx.x * 256 + threadIdx.x;
    int e    = gid >> 5;
    int lane = gid & 31;
    if (e >= E_EDGES) return;
    int s = src[e], d = dst[e];
    int h = lane >> 2;                       // 8 contiguous dims per lane -> one head
    float alpha = exbuf[(size_t)e * HEADS + h] / denom[(size_t)d * HEADS + h];
    const float4* zp = (const float4*)(z + (size_t)s * D_HID + lane * 8);
    float4 z0 = zp[0], z1 = zp[1];
    float* ap = acc + (size_t)d * D_HID + lane * 8;
    atomic_add_f32(ap + 0, z0.x * alpha); atomic_add_f32(ap + 1, z0.y * alpha);
    atomic_add_f32(ap + 2, z0.z * alpha); atomic_add_f32(ap + 3, z0.w * alpha);
    atomic_add_f32(ap + 4, z1.x * alpha); atomic_add_f32(ap + 5, z1.y * alpha);
    atomic_add_f32(ap + 6, z1.z * alpha); atomic_add_f32(ap + 7, z1.w * alpha);
}

__global__ void k_relu_bf16(const float* __restrict__ a, __bf16* __restrict__ o, int n) {
    int i = blockIdx.x * 256 + threadIdx.x;
    if (i < n) { float v = a[i]; o[i] = (__bf16)(v > 0.f ? v : 0.f); }
}

// ---------------- semantic attention: score += sum_rows q . tanh(out @ Wk + bk) ----------------
__global__ __launch_bounds__(32)
void k_semantic_score(const __bf16* __restrict__ A, const __bf16* __restrict__ WkT,
                      const float* __restrict__ bk, const float* __restrict__ q,
                      float* __restrict__ score) {
    const int K    = D_HID;
    const int m0   = blockIdx.x << 4;
    const int n0   = blockIdx.y * 64;        // NT = 4
    const int lane = threadIdx.x;
    const int half = lane >> 4;
    const int l15  = lane & 15;
    const __bf16* arow = A   + (size_t)(m0 + l15) * K;
    const __bf16* brow = WkT + (size_t)(n0 + l15) * K;
    v8f acc[4];
#pragma unroll
    for (int t = 0; t < 4; t++) acc[t] = (v8f){};
    wmma_panel<4>(arow, brow, K, half, acc);
    float part = 0.f;
#pragma unroll
    for (int t = 0; t < 4; t++) {
        const int col = n0 + t * 16 + l15;
        const float bv = bk[col], qv = q[col];
        float s = 0.f;
#pragma unroll
        for (int v = 0; v < 8; v++) s += tanhf(acc[t][v] + bv);
        part += s * qv;
    }
#pragma unroll
    for (int off = 16; off > 0; off >>= 1) part += __shfl_down(part, off, 32);
    if (lane == 0) atomic_add_f32(score, part);
}

__global__ void k_beta(const float* __restrict__ score, float* __restrict__ beta) {
    if (threadIdx.x == 0 && blockIdx.x == 0) {
        float s[P_TYPES]; float m = -1e30f;
        for (int p = 0; p < P_TYPES; p++) { s[p] = score[p] * (1.0f / N_NODES); m = fmaxf(m, s[p]); }
        float sum = 0.f;
        for (int p = 0; p < P_TYPES; p++) { s[p] = __expf(s[p] - m); sum += s[p]; }
        for (int p = 0; p < P_TYPES; p++) beta[p] = s[p] / sum;
    }
}

// h = act( sum_p beta_p * out_p ), stored bf16 for the next GEMM
__global__ void k_combine(const __bf16* __restrict__ outbf, const float* __restrict__ beta,
                          __bf16* __restrict__ hbf, int do_elu) {
    int i = blockIdx.x * 256 + threadIdx.x;
    if (i >= N_NODES * D_HID) return;
    float v = 0.f;
#pragma unroll
    for (int p = 0; p < P_TYPES; p++)
        v += beta[p] * (float)outbf[(size_t)p * N_NODES * D_HID + i];
    if (do_elu) v = (v > 0.f) ? v : (__expf(v) - 1.f);
    hbf[i] = (__bf16)v;
}

// ---------------- one full HAN layer ----------------
static void run_layer(const __bf16* in_bf, int K_in, const __bf16* WpT, const float* bp,
                      const float* att_s, const float* att_d, const __bf16* WkT,
                      const float* bk, const float* q, const int* esrc, const int* edst,
                      float* z, float* a_src, float* a_dst, float* denom, float* exbuf,
                      float* acc, __bf16* outbf, float* score, float* beta,
                      __bf16* hbf, int do_elu, hipStream_t stream) {
    // z = in @ Wp + bp   (16x64 tile per wave)
    k_wmma_gemm_bias<4><<<dim3(N_NODES / 16, D_HID / 64), 32, 0, stream>>>(in_bf, WpT, bp, z,
                                                                           K_in, D_HID);
    // per-head src/dst coefficients
    k_head_scores<<<(N_NODES * HEADS) / 256, 256, 0, stream>>>(z, att_s, att_d, a_src, a_dst);
    k_zero_f32<<<1, 256, 0, stream>>>(score, P_TYPES);
    for (int p = 0; p < P_TYPES; p++) {
        const int* sp = esrc + (size_t)p * E_EDGES;
        const int* dp = edst + (size_t)p * E_EDGES;
        k_zero_f32<<<(N_NODES * HEADS) / 256, 256, 0, stream>>>(denom, N_NODES * HEADS);
        k_zero_f32<<<(N_NODES * D_HID) / 256, 256, 0, stream>>>(acc, N_NODES * D_HID);
        k_edge_softmax_denom<<<E_EDGES / 256, 256, 0, stream>>>(sp, dp, a_src, a_dst, exbuf, denom);
        k_edge_scatter<<<(E_EDGES * 32) / 256, 256, 0, stream>>>(sp, dp, z, exbuf, denom, acc);
        k_relu_bf16<<<(N_NODES * D_HID) / 256, 256, 0, stream>>>(acc, outbf + (size_t)p * N_NODES * D_HID,
                                                                 N_NODES * D_HID);
        k_semantic_score<<<dim3(N_NODES / 16, D_HID / 64), 32, 0, stream>>>(
            outbf + (size_t)p * N_NODES * D_HID, WkT, bk, q, score + p);
    }
    k_beta<<<1, 32, 0, stream>>>(score, beta);
    k_combine<<<(N_NODES * D_HID) / 256, 256, 0, stream>>>(outbf, beta, hbf, do_elu);
}

extern "C" void kernel_launch(void* const* d_in, const int* in_sizes, int n_in,
                              void* d_out, int out_size, void* d_ws, size_t ws_size,
                              hipStream_t stream) {
    const float* x    = (const float*)d_in[0];
    const int*   esrc = (const int*)d_in[1];
    const int*   edst = (const int*)d_in[2];
    const float* Wp1  = (const float*)d_in[3];
    const float* bp1  = (const float*)d_in[4];
    const float* as1  = (const float*)d_in[5];
    const float* ad1  = (const float*)d_in[6];
    const float* Wk1  = (const float*)d_in[7];
    const float* bk1  = (const float*)d_in[8];
    const float* q1   = (const float*)d_in[9];
    const float* Wp2  = (const float*)d_in[10];
    const float* bp2  = (const float*)d_in[11];
    const float* as2  = (const float*)d_in[12];
    const float* ad2  = (const float*)d_in[13];
    const float* Wk2  = (const float*)d_in[14];
    const float* bk2  = (const float*)d_in[15];
    const float* q2   = (const float*)d_in[16];
    const float* Wo   = (const float*)d_in[17];
    const float* bo   = (const float*)d_in[18];
    float* out = (float*)d_out;

    char* cur = (char*)d_ws;
    auto carve = [&](size_t bytes) {
        char* p = cur;
        cur += (bytes + 255) & ~(size_t)255;
        return p;
    };
    __bf16* x_bf  = (__bf16*)carve((size_t)N_NODES * IN_DIM * 2);
    __bf16* WpT1  = (__bf16*)carve((size_t)D_HID * IN_DIM * 2);
    __bf16* WkT1  = (__bf16*)carve((size_t)D_HID * D_HID * 2);
    __bf16* WpT2  = (__bf16*)carve((size_t)D_HID * D_HID * 2);
    __bf16* WkT2  = (__bf16*)carve((size_t)D_HID * D_HID * 2);
    __bf16* WoT   = (__bf16*)carve((size_t)CLASSES * D_HID * 2);
    float*  z     = (float*)carve((size_t)N_NODES * D_HID * 4);
    float*  a_src = (float*)carve((size_t)N_NODES * HEADS * 4);
    float*  a_dst = (float*)carve((size_t)N_NODES * HEADS * 4);
    float*  denom = (float*)carve((size_t)N_NODES * HEADS * 4);
    float*  exbuf = (float*)carve((size_t)E_EDGES * HEADS * 4);
    float*  acc   = (float*)carve((size_t)N_NODES * D_HID * 4);
    __bf16* outbf = (__bf16*)carve((size_t)P_TYPES * N_NODES * D_HID * 2);
    __bf16* hbf   = (__bf16*)carve((size_t)N_NODES * D_HID * 2);
    float*  score = (float*)carve(256);
    float*  beta  = (float*)carve(256);

    // ---- weight / input preparation (deterministic, every call) ----
    k_f32_to_bf16<<<(N_NODES * IN_DIM) / 256, 256, 0, stream>>>(x, x_bf, N_NODES * IN_DIM);
    k_transpose_to_bf16<<<(IN_DIM * D_HID + 255) / 256, 256, 0, stream>>>(Wp1, WpT1, IN_DIM, D_HID);
    k_transpose_to_bf16<<<(D_HID * D_HID + 255) / 256, 256, 0, stream>>>(Wk1, WkT1, D_HID, D_HID);
    k_transpose_to_bf16<<<(D_HID * D_HID + 255) / 256, 256, 0, stream>>>(Wp2, WpT2, D_HID, D_HID);
    k_transpose_to_bf16<<<(D_HID * D_HID + 255) / 256, 256, 0, stream>>>(Wk2, WkT2, D_HID, D_HID);
    k_transpose_to_bf16<<<(D_HID * CLASSES + 255) / 256, 256, 0, stream>>>(Wo, WoT, D_HID, CLASSES);

    // ---- layer 1 (input = x_bf, K = 512), output h (elu) -> hbf ----
    run_layer(x_bf, IN_DIM, WpT1, bp1, as1, ad1, WkT1, bk1, q1, esrc, edst,
              z, a_src, a_dst, denom, exbuf, acc, outbf, score, beta, hbf, /*elu=*/1, stream);

    // ---- layer 2 (input = hbf, K = 256), output h2 -> hbf ----
    run_layer(hbf, D_HID, WpT2, bp2, as2, ad2, WkT2, bk2, q2, esrc, edst,
              z, a_src, a_dst, denom, exbuf, acc, outbf, score, beta, hbf, /*elu=*/0, stream);

    // ---- classifier: out = h2 @ Wo + bo (single 16-wide N tile) ----
    k_wmma_gemm_bias<1><<<dim3(N_NODES / 16, CLASSES / 16), 32, 0, stream>>>(hbf, WoT, bo, out,
                                                                             D_HID, CLASSES);
}
